// SpatialAttention_16819091931213
// MI455X (gfx1250) — matrix operand order repaired and checked by reading, not compiled, not run on hardware
//
#include <hip/hip_runtime.h>
#include <hip/hip_bf16.h>
#include <cstdint>

// ---------------------------------------------------------------------------
// SpatialAttention for MI455X (gfx1250, wave32, WMMA).
// - All GEMMs: v_wmma_f32_16x16x32_f16 (fp32 accumulate).
// - Flash-attention tiling: 1024x1024 score matrix never touches HBM.
// - Weight/A tiles: async global->LDS DMA (ASYNCcnt-tracked).
// - Streaming operands prefetched one K-step ahead (global_prefetch_b8).
// ---------------------------------------------------------------------------

typedef __attribute__((ext_vector_type(16))) _Float16 v16h;
typedef __attribute__((ext_vector_type(8)))  float    v8f;
typedef int v4i_vs __attribute__((vector_size(16)));   // matches builtin param

#define ALD 40   // padded LDS row (f16 elems) for 32-wide tiles (80B = 16B*5)
#define PLD 72   // padded LDS row for 64-wide P tile (144B = 16B*9)
#define SLD 68   // padded LDS row (f32) for 64-wide score tile

// ---- async global -> LDS copy (16B per lane), ASYNCcnt-tracked -------------
__device__ __forceinline__ void async_g2l_b128(void* lds, const void* g) {
#if __has_builtin(__builtin_amdgcn_global_load_async_to_lds_b128)
    __builtin_amdgcn_global_load_async_to_lds_b128(
        (__attribute__((address_space(1))) v4i_vs*)(unsigned long long)g,
        (__attribute__((address_space(3))) v4i_vs*)(unsigned int)(unsigned long long)lds,
        0, 0);
#else
    asm volatile("global_load_async_to_lds_b128 %0, %1, off"
                 :: "v"((unsigned int)(unsigned long long)lds),
                    "v"((unsigned long long)g)
                 : "memory");
#endif
}

__device__ __forceinline__ void wait_async0() {
#if __has_builtin(__builtin_amdgcn_s_wait_asynccnt)
    __builtin_amdgcn_s_wait_asynccnt(0);
#else
    asm volatile("s_wait_asynccnt 0x0" ::: "memory");
#endif
}

// ---- WMMA fragment loaders (contiguous k = 16*(lane>>4) + elem, wave32) ----------

// A 16x32 f16, row-major source (row stride ld f16 elems).
// lane m=lane&15 holds row M=m; elems 0..15 = K hf*16 + 0..15
__device__ __forceinline__ v16h load_a_frag(const _Float16* base, int ld) {
    const int lane = threadIdx.x & 31;
    const int m  = lane & 15;
    const int hf = lane >> 4;
    const _Float16* p = base + m * ld;
    union { v16h v; uint4 q[2]; } t;
    t.q[0] = *(const uint4*)(p + hf * 16);
    t.q[1] = *(const uint4*)(p + hf * 16 + 8);
    return t.v;
}

// B 32x16 f16 from a column-major tile Bt[n][k] (row stride ld f16 elems).
// lane n=lane&15 holds col N=n; elems 0..15 = K (lane>>4)*16 + 0..15
__device__ __forceinline__ v16h load_b_frag(const _Float16* base, int ld) {
    const int lane = threadIdx.x & 31;
    const int n  = lane & 15;
    const int hf = lane >> 4;
    const _Float16* p = base + n * ld + hf * 16;
    union { v16h v; uint4 q[2]; } t;
    t.q[0] = *(const uint4*)(p);
    t.q[1] = *(const uint4*)(p + 8);
    return t.v;
}

// ---- 1) weight conversion fp32 -> f16 --------------------------------------
__global__ void k_convert_w(const float* __restrict__ qkv_w,
                            const float* __restrict__ proj_w,
                            _Float16* __restrict__ qw,
                            _Float16* __restrict__ pw) {
    int i = blockIdx.x * 256 + threadIdx.x;
    if (i < 768 * 256) qw[i] = (_Float16)qkv_w[i];
    if (i < 256 * 256) pw[i] = (_Float16)proj_w[i];
}

// ---- 2) GroupNorm: block per (n, group); 8 channels x 1024 spatial ---------
__global__ void k_groupnorm(const float* __restrict__ x,
                            const float* __restrict__ gn_w,
                            const float* __restrict__ gn_b,
                            _Float16* __restrict__ xn) {
    const int n  = blockIdx.x >> 5;          // 0..15 (= bi*8 + ti)
    const int g  = blockIdx.x & 31;          // group
    const int bi = n >> 3, ti = n & 7;
    const int tid = threadIdx.x;
    float s0 = 0.f, s1 = 0.f;
    for (int i = 0; i < 32; ++i) {
        int e  = tid + i * 256;              // 0..8191
        int cl = e >> 10, sp = e & 1023;
        int c  = g * 8 + cl;
        float v = x[(((size_t)bi * 256 + c) * 8 + ti) * 1024 + sp];
        s0 += v; s1 += v * v;
    }
    __shared__ float r0[256], r1[256];
    r0[tid] = s0; r1[tid] = s1;
    __syncthreads();
    for (int off = 128; off > 0; off >>= 1) {
        if (tid < off) { r0[tid] += r0[tid + off]; r1[tid] += r1[tid + off]; }
        __syncthreads();
    }
    __shared__ float s_mean, s_inv;
    if (tid == 0) {
        float m   = r0[0] * (1.0f / 8192.0f);
        float var = r1[0] * (1.0f / 8192.0f) - m * m;
        s_mean = m;
        s_inv  = rsqrtf(var + 1e-5f);
    }
    __syncthreads();
    const float mean = s_mean, inv = s_inv;
    for (int i = 0; i < 32; ++i) {
        int e  = tid + i * 256;
        int cl = e >> 10, sp = e & 1023;
        int c  = g * 8 + cl;
        float v = x[(((size_t)bi * 256 + c) * 8 + ti) * 1024 + sp];
        float y = (v - mean) * inv * gn_w[c] + gn_b[c];
        xn[((size_t)n * 256 + c) * 1024 + sp] = (_Float16)y;
    }
}

// ---- 3) QKV GEMM: out[n,o,s] = sum_c W[o,c]*xn[n,c,s] + b[o]  (f16 out) ----
// grid (12*16, 16); block 256 = 8 waves; 64x64 tile; K steps of 32.
__global__ void k_qkv_gemm(const _Float16* __restrict__ W,
                           const _Float16* __restrict__ X,
                           const float* __restrict__ bias,
                           _Float16* __restrict__ out) {
    const int n  = blockIdx.y;
    const int mt = blockIdx.x >> 4;
    const int nt = blockIdx.x & 15;
    const int Mb = mt * 64, Nb = nt * 64;
    const _Float16* Xn = X + (size_t)n * 256 * 1024;
    __shared__ alignas(16) _Float16 As[64][ALD];   // A row-major [m][k]
    __shared__ alignas(16) _Float16 Bs[64][ALD];   // B col-major [nn][kk]
    const int tid = threadIdx.x;
    const int w   = tid >> 5;
    const int tm  = w >> 1;
    const int tnb = (w & 1) * 2;
    // per-thread copy slots (hoisted)
    const int am  = tid >> 2, ak = (tid & 3) * 8;      // A: 16B @ [am][ak]
    const int kk  = tid >> 3, nn0 = (tid & 7) * 8;     // B: 16B run of nn
    v8f acc0 = {}; v8f acc1 = {};
    for (int kb = 0; kb < 8; ++kb) {
        // A tile: async DMA global->LDS, no VGPR staging
        async_g2l_b128(&As[am][ak],
                       W + (size_t)(Mb + am) * 256 + kb * 32 + ak);
        {   // B: Bs[nn][kk] = X[kb*32+kk][Nb+nn]  (transpose on store)
            union { uint4 q; _Float16 h[8]; } v;
            v.q = *(const uint4*)(Xn + (size_t)(kb * 32 + kk) * 1024 + Nb + nn0);
#pragma unroll
            for (int j = 0; j < 8; ++j) Bs[nn0 + j][kk] = v.h[j];
        }
        if (kb < 7) {   // prefetch next K-step operands into WGP$/L2
            __builtin_prefetch(W + (size_t)(Mb + am) * 256 + (kb + 1) * 32 + ak, 0, 0);
            __builtin_prefetch(Xn + (size_t)((kb + 1) * 32 + kk) * 1024 + Nb + nn0, 0, 0);
        }
        wait_async0();
        __syncthreads();
        v16h a  = load_a_frag(&As[tm * 16][0], ALD);
        v16h b0 = load_b_frag(&Bs[tnb * 16][0], ALD);
        v16h b1 = load_b_frag(&Bs[(tnb + 1) * 16][0], ALD);
        acc0 = __builtin_amdgcn_wmma_f32_16x16x32_f16(false, a, false, b0, (short)0, acc0, false, false);
        acc1 = __builtin_amdgcn_wmma_f32_16x16x32_f16(false, a, false, b1, (short)0, acc1, false, false);
        __syncthreads();
    }
    const int lane = tid & 31;
    const int col = lane & 15, hf = lane >> 4;
    _Float16* On = out + (size_t)n * 768 * 1024;
#pragma unroll
    for (int r = 0; r < 8; ++r) {
        int o  = Mb + tm * 16 + hf * 8 + r;
        float b = bias[o];
        On[(size_t)o * 1024 + Nb + tnb * 16 + col]       = (_Float16)(acc0[r] + b);
        On[(size_t)o * 1024 + Nb + (tnb + 1) * 16 + col] = (_Float16)(acc1[r] + b);
    }
}

// ---- 4) Flash attention: block per (n, head, 64-row q tile) ---------------
// qkv layout [n][768][1024] f16: q rows h*32+d, k rows 256+h*32+d, v 512+...
__global__ void k_attn(const _Float16* __restrict__ qkv,
                       _Float16* __restrict__ attn_out) {
    const int n  = blockIdx.y;
    const int qt = blockIdx.x & 15;
    const int hh = blockIdx.x >> 4;
    const int tid = threadIdx.x;
    const int w = tid >> 5;
    const int lane = tid & 31;
    const int col = lane & 15, hf = lane >> 4;

    __shared__ alignas(16) _Float16 Qs[64][ALD];
    __shared__ alignas(16) _Float16 Ks[64][ALD];
    __shared__ alignas(16) float    Sc[64][SLD];
    __shared__ alignas(16) _Float16 Ps[64][PLD];
    __shared__ float mstate[64], lstate[64], alphas[64];

    const _Float16* qbase = qkv + ((size_t)n * 768 + hh * 32) * 1024;
    const _Float16* kbase = qkv + ((size_t)n * 768 + 256 + hh * 32) * 1024;
    const _Float16* vbase = qkv + ((size_t)n * 768 + 512 + hh * 32) * 1024;

    const int d = tid >> 3, m0 = (tid & 7) * 8;    // tile-copy slot
    {   // Q tile: Qs[m][d] = qbase[d*1024 + qt*64 + m]
        union { uint4 q; _Float16 h[8]; } v;
        v.q = *(const uint4*)(qbase + (size_t)d * 1024 + qt * 64 + m0);
#pragma unroll
        for (int j = 0; j < 8; ++j) Qs[m0 + j][d] = v.h[j];
    }
    if (tid < 64) { mstate[tid] = -1e30f; lstate[tid] = 0.f; }
    __syncthreads();

    const int stm = w >> 1, stnb = (w & 1) * 2;  // score-tile assignment (2/wave)
    const int otm = w >> 1, otn  = w & 1;        // O-tile assignment (1/wave)
    v8f oacc = {};
    const float scale = 0.17677669529663687f;    // 1/sqrt(32)

    for (int kb = 0; kb < 16; ++kb) {
        {   // K tile: Ks[nk][d] = kbase[d*1024 + kb*64 + nk]
            union { uint4 q; _Float16 h[8]; } v;
            v.q = *(const uint4*)(kbase + (size_t)d * 1024 + kb * 64 + m0);
#pragma unroll
            for (int j = 0; j < 8; ++j) Ks[m0 + j][d] = v.h[j];
        }
        if (kb < 15) {  // prefetch next K tile and V block
            __builtin_prefetch(kbase + (size_t)d * 1024 + (kb + 1) * 64 + m0, 0, 0);
            __builtin_prefetch(vbase + (size_t)d * 1024 + (kb + 1) * 64 + m0, 0, 0);
        }
        __syncthreads();
        {   // S = scale * Q K^T (two 16x16 tiles per wave)
            v16h a  = load_a_frag(&Qs[stm * 16][0], ALD);
            v16h b0 = load_b_frag(&Ks[stnb * 16][0], ALD);
            v16h b1 = load_b_frag(&Ks[(stnb + 1) * 16][0], ALD);
            v8f s0 = {}; v8f s1 = {};
            s0 = __builtin_amdgcn_wmma_f32_16x16x32_f16(false, a, false, b0, (short)0, s0, false, false);
            s1 = __builtin_amdgcn_wmma_f32_16x16x32_f16(false, a, false, b1, (short)0, s1, false, false);
#pragma unroll
            for (int r = 0; r < 8; ++r) {
                int row = stm * 16 + hf * 8 + r;
                Sc[row][stnb * 16 + col]       = s0[r] * scale;
                Sc[row][(stnb + 1) * 16 + col] = s1[r] * scale;
            }
        }
        __syncthreads();
        if (tid < 64) {  // online softmax, one thread per q row
            float mprev = mstate[tid];
            float rmax = mprev;
            for (int c = 0; c < 64; ++c) rmax = fmaxf(rmax, Sc[tid][c]);
            float a_ = __expf(mprev - rmax);
            float ls = 0.f;
            for (int c = 0; c < 64; ++c) {
                float p = __expf(Sc[tid][c] - rmax);
                Ps[tid][c] = (_Float16)p;
                ls += p;
            }
            lstate[tid] = lstate[tid] * a_ + ls;
            mstate[tid] = rmax;
            alphas[tid] = a_;
        }
        __syncthreads();
        {   // O = alpha*O + P V   (K=64 -> two WMMAs; V fragment from global)
#pragma unroll
            for (int r = 0; r < 8; ++r) oacc[r] *= alphas[otm * 16 + hf * 8 + r];
#pragma unroll
            for (int kc = 0; kc < 2; ++kc) {
                v16h a = load_a_frag(&Ps[otm * 16][kc * 32], PLD);
                const _Float16* vp = vbase + (size_t)(otn * 16 + col) * 1024
                                     + kb * 64 + kc * 32 + hf * 16;
                union { v16h v; uint4 q[2]; } t;
                t.q[0] = *(const uint4*)(vp);
                t.q[1] = *(const uint4*)(vp + 8);
                oacc = __builtin_amdgcn_wmma_f32_16x16x32_f16(false, a, false, t.v, (short)0, oacc, false, false);
            }
        }
        __syncthreads();
    }
    {   // epilogue: O /= l, store transposed to [c][s] layout, 16B per lane
        union { uint4 q; _Float16 h[8]; } pk;
#pragma unroll
        for (int r = 0; r < 8; ++r) {
            int row = otm * 16 + hf * 8 + r;
            pk.h[r] = (_Float16)(oacc[r] / lstate[row]);
        }
        _Float16* op = attn_out
            + ((size_t)n * 256 + hh * 32 + otn * 16 + col) * 1024
            + qt * 64 + otm * 16 + hf * 8;
        *(uint4*)op = pk.q;
    }
}

// ---- 5) Proj GEMM + bias + residual, fp32 out in [b,c,t,h,w] --------------
__global__ void k_proj_gemm(const _Float16* __restrict__ W,
                            const _Float16* __restrict__ X,
                            const float* __restrict__ bias,
                            const float* __restrict__ xres,
                            float* __restrict__ out) {
    const int n  = blockIdx.y;
    const int mt = blockIdx.x >> 4;
    const int nt = blockIdx.x & 15;
    const int Mb = mt * 64, Nb = nt * 64;
    const int bi = n >> 3, ti = n & 7;
    const _Float16* Xn = X + (size_t)n * 256 * 1024;
    __shared__ alignas(16) _Float16 As[64][ALD];
    __shared__ alignas(16) _Float16 Bs[64][ALD];
    const int tid = threadIdx.x;
    const int w   = tid >> 5;
    const int tm  = w >> 1;
    const int tnb = (w & 1) * 2;
    const int am  = tid >> 2, ak = (tid & 3) * 8;
    const int kk  = tid >> 3, nn0 = (tid & 7) * 8;
    v8f acc0 = {}; v8f acc1 = {};
    for (int kb = 0; kb < 8; ++kb) {
        async_g2l_b128(&As[am][ak],
                       W + (size_t)(Mb + am) * 256 + kb * 32 + ak);
        {
            union { uint4 q; _Float16 h[8]; } v;
            v.q = *(const uint4*)(Xn + (size_t)(kb * 32 + kk) * 1024 + Nb + nn0);
#pragma unroll
            for (int j = 0; j < 8; ++j) Bs[nn0 + j][kk] = v.h[j];
        }
        if (kb < 7) {
            __builtin_prefetch(W + (size_t)(Mb + am) * 256 + (kb + 1) * 32 + ak, 0, 0);
            __builtin_prefetch(Xn + (size_t)((kb + 1) * 32 + kk) * 1024 + Nb + nn0, 0, 0);
        }
        wait_async0();
        __syncthreads();
        v16h a  = load_a_frag(&As[tm * 16][0], ALD);
        v16h b0 = load_b_frag(&Bs[tnb * 16][0], ALD);
        v16h b1 = load_b_frag(&Bs[(tnb + 1) * 16][0], ALD);
        acc0 = __builtin_amdgcn_wmma_f32_16x16x32_f16(false, a, false, b0, (short)0, acc0, false, false);
        acc1 = __builtin_amdgcn_wmma_f32_16x16x32_f16(false, a, false, b1, (short)0, acc1, false, false);
        __syncthreads();
    }
    const int lane = tid & 31;
    const int col = lane & 15, hf = lane >> 4;
#pragma unroll
    for (int r = 0; r < 8; ++r) {
        int o  = Mb + tm * 16 + hf * 8 + r;
        float b = bias[o];
        int s0 = Nb + tnb * 16 + col;
        int s1 = Nb + (tnb + 1) * 16 + col;
        size_t i0 = (((size_t)bi * 256 + o) * 8 + ti) * 1024 + s0;
        size_t i1 = (((size_t)bi * 256 + o) * 8 + ti) * 1024 + s1;
        out[i0] = xres[i0] + acc0[r] + b;
        out[i1] = xres[i1] + acc1[r] + b;
    }
}

// ---------------------------------------------------------------------------
extern "C" void kernel_launch(void* const* d_in, const int* in_sizes, int n_in,
                              void* d_out, int out_size, void* d_ws, size_t ws_size,
                              hipStream_t stream) {
    (void)in_sizes; (void)n_in; (void)out_size; (void)ws_size;
    const float* x      = (const float*)d_in[0];
    const float* gn_w   = (const float*)d_in[1];
    const float* gn_b   = (const float*)d_in[2];
    const float* qkv_w  = (const float*)d_in[3];
    const float* qkv_b  = (const float*)d_in[4];
    const float* proj_w = (const float*)d_in[5];
    const float* proj_b = (const float*)d_in[6];
    float* out = (float*)d_out;

    char* ws = (char*)d_ws;                      // workspace layout (bytes):
    _Float16* qw   = (_Float16*)(ws + 0);        //  768*256 f16
    _Float16* pw   = (_Float16*)(ws + 393216);   //  256*256 f16
    _Float16* xn   = (_Float16*)(ws + 524288);   //  16*256*1024 f16
    _Float16* qkv  = (_Float16*)(ws + 8912896);  //  16*768*1024 f16
    _Float16* attn = (_Float16*)(ws + 34078720); //  16*256*1024 f16  (end ~42.5MB)

    k_convert_w<<<768, 256, 0, stream>>>(qkv_w, proj_w, qw, pw);
    k_groupnorm<<<512, 256, 0, stream>>>(x, gn_w, gn_b, xn);
    k_qkv_gemm<<<dim3(192, 16), 256, 0, stream>>>(qw, xn, qkv_b, qkv);
    k_attn<<<dim3(128, 16), 256, 0, stream>>>(qkv, attn);
    k_proj_gemm<<<dim3(64, 16), 256, 0, stream>>>(pw, attn, proj_b, x, out);
}
